// GAT_53618371723423
// MI455X (gfx1250) — matrix operand order repaired and checked by reading, not compiled, not run on hardware
//
#include <hip/hip_runtime.h>
#include <hip/hip_bf16.h>
#include <math.h>

// ---------------------------------------------------------------------------
// GAT 2-layer forward for gfx1250 (MI455X).
//  - Inputs pre-converted fp32 -> bf16 once; GEMMs via
//    v_wmma_f32_16x16x32_bf16 with B staged in LDS in fragment layout
//    (branch-free inner loop: 2x global b128 + 2*NT ds b128 + NT wmma).
//  - dst sorted => CSR rowptr, per-dst-node fused segment softmax +
//    aggregation (no atomics, deterministic, graph-capture safe).
// ---------------------------------------------------------------------------

typedef __attribute__((ext_vector_type(16))) __bf16 v16bf;
typedef __attribute__((ext_vector_type(8)))  __bf16 v8bf;
typedef __attribute__((ext_vector_type(4)))  __bf16 v4bf;
typedef __attribute__((ext_vector_type(8)))  float  v8f;
typedef __attribute__((ext_vector_type(4)))  float  v4f;

#define NEG_SLOPE 0.2f

__device__ __forceinline__ float leaky(float v) {
  return v > 0.f ? v : NEG_SLOPE * v;
}

// ---------------------------------------------------------------------------
// fp32 -> bf16 (RNE), 4 elements/thread. n must be a multiple of 4
// (12.8M, 65536, 12032 here -- all are).
// ---------------------------------------------------------------------------
__global__ void f32_to_bf16_kernel(const float* __restrict__ in,
                                   __bf16* __restrict__ out, int n4) {
  const int i = blockIdx.x * blockDim.x + threadIdx.x;
  if (i >= n4) return;
  const v4f v = ((const v4f*)in)[i];
  v4bf o;
#pragma unroll
  for (int e = 0; e < 4; ++e) o[e] = (__bf16)v[e];
  ((v4bf*)out)[i] = o;
}

// ---------------------------------------------------------------------------
// WMMA GEMM: out[M,N] = A[M,K] * W[K,N]  (bf16 in, fp32 out, row-major).
// Block = 256 threads = 8 waves. All waves share one N-group of NT*16 cols;
// wave w owns rows [(,blockIdx.x*8+w)*16, +16). B for the whole K range is
// staged once into LDS in exact WMMA B-fragment layout (zero-padded for
// ragged N), so the K loop is branch-free.
//   A frag (16x32 bf16): lanes 0-15 M=lane, elems 0-7 K 0-7, 8-15 K 16-23;
//                        lanes16-31 M=lane-16, K 8-15 / 24-31.
//   B frag (32x16): col = lane&15, elems 0-7 K 8*(lane>>4)+e, 8-15 K 16+8*(lane>>4)+(e-8) (as A).
//   C: VGPR r -> row r + 8*(lane>=16), col = lane&15.
// Requires K % 32 == 0 (K=256 here). Dynamic LDS = NT*(K/32)*1024 bytes.
// ---------------------------------------------------------------------------
template <int NT>
__global__ __launch_bounds__(256) void gemm_wmma_bf16(
    const __bf16* __restrict__ A, const __bf16* __restrict__ W,
    float* __restrict__ out, int M, int K, int N) {
  extern __shared__ __bf16 smemB[];  // [NF][32 lanes][16 bf16]
  const int KF = K >> 5;             // k-fragments
  const int NF = NT * KF;
  const int n0base = blockIdx.y * 16 * NT;

  // ---- stage B into LDS in fragment layout (once per block) ----
  for (int slot = threadIdx.x; slot < NF * 32; slot += blockDim.x) {
    const int f = slot >> 5;          // fragment = t*KF + kk
    const int l = slot & 31;          // lane within fragment
    const int t = f / KF;
    const int kk = f % KF;
    const int col = n0base + t * 16 + (l & 15);
    const int kbase = (kk << 5) + ((l >> 4) << 3);
    __bf16* dstp = smemB + (size_t)slot * 16;
    if (col < N) {
#pragma unroll
      for (int e = 0; e < 16; ++e)
        dstp[e] = W[(size_t)(kbase + (e & 7) + ((e >> 3) << 4)) * N + col];
    } else {
#pragma unroll
      for (int e = 0; e < 16; ++e) dstp[e] = (__bf16)0.0f;
    }
  }
  __syncthreads();

  // ---- per-wave 16-row stripe ----
  const int wave = threadIdx.x >> 5;
  const int lane = threadIdx.x & 31;
  const int row0 = (blockIdx.x * (blockDim.x >> 5) + wave) * 16;
  if (row0 >= M) return;  // wave-uniform; no barriers after this point

  const int m    = lane & 15;
  const int koff = (lane >> 4) << 3;   // 0 or 8 elements
  const __bf16* arow = A + (size_t)(row0 + m) * K;

  v8f acc[NT] = {};
  for (int k0 = 0; k0 < K; k0 += 32) {
    // two aligned 16B chunks -> one v16bf A fragment
    const v8bf alo = *(const v8bf*)(arow + k0 + koff);
    const v8bf ahi = *(const v8bf*)(arow + k0 + 16 + koff);
    const v16bf a = __builtin_shufflevector(
        alo, ahi, 0, 1, 2, 3, 4, 5, 6, 7, 8, 9, 10, 11, 12, 13, 14, 15);
    const int kk = k0 >> 5;
#pragma unroll
    for (int t = 0; t < NT; ++t) {
      const v16bf b = *(const v16bf*)(
          smemB + ((size_t)(t * KF + kk) * 32 + lane) * 16);
      acc[t] = __builtin_amdgcn_wmma_f32_16x16x32_bf16(
          false, a, false, b, (short)0, acc[t], false, false);
    }
  }

  const int ncol  = lane & 15;
  const int rowhi = (lane >> 4) << 3;
#pragma unroll
  for (int t = 0; t < NT; ++t) {
    const int c = n0base + t * 16 + ncol;
    if (c < N) {
#pragma unroll
      for (int r = 0; r < 8; ++r)
        out[(size_t)(row0 + r + rowhi) * N + c] = acc[t][r];
    }
  }
}

// ---------------------------------------------------------------------------
// Per-node attention logits: el[n,h] = fs[n,h,:]·al[h,:], er likewise.
// One thread per (node, head); fs row offset == idx*D since idx = n*H+h.
// ---------------------------------------------------------------------------
__global__ void attn_logits_kernel(const float* __restrict__ fs,
                                   const float* __restrict__ al,
                                   const float* __restrict__ ar,
                                   float* __restrict__ el,
                                   float* __restrict__ er,
                                   int nNodes, int H, int D) {
  const int idx = blockIdx.x * blockDim.x + threadIdx.x;
  if (idx >= nNodes * H) return;
  const int h = idx % H;
  const float* f = fs + (size_t)idx * D;
  float sl = 0.f, sr = 0.f;
  for (int d = 0; d < D; ++d) {
    const float v = f[d];
    sl = fmaf(v, al[h * D + d], sl);
    sr = fmaf(v, ar[h * D + d], sr);
  }
  el[idx] = sl;
  er[idx] = sr;
}

// ---------------------------------------------------------------------------
// Build CSR rowptr from sorted dst (rowptr has nNodes+1 entries).
// ---------------------------------------------------------------------------
__global__ void build_rowptr_kernel(const int* __restrict__ dst,
                                    int* __restrict__ rowptr,
                                    int E, int nNodes) {
  const int i = blockIdx.x * blockDim.x + threadIdx.x;
  if (i >= E) return;
  const int d     = dst[i];
  const int dprev = (i == 0) ? -1 : dst[i - 1];
  for (int n = dprev + 1; n <= d; ++n) rowptr[n] = i;
  if (i == E - 1) {
    for (int n = d + 1; n <= nNodes; ++n) rowptr[n] = E;
  }
}

// ---------------------------------------------------------------------------
// Fused segment softmax + weighted aggregation, one dst node per block.
//   out[node, h*D+d] = sum_e alpha[e,h] * fs[src_e, h*D+d] + bias[h*D+d]
// Thread t owns channel c=t. Alphas for 64-edge chunks staged in LDS; the
// gather of fs[src] rows is fully coalesced (1 float/lane). Zero in-degree
// -> out = bias (segment_sum identity). OutT lets layer-1 emit bf16
// activations directly for the layer-2 WMMA GEMM.
// ---------------------------------------------------------------------------
template <int H, int D, bool RELU, typename OutT>
__global__ void gat_aggregate(const float* __restrict__ fs,
                              const float* __restrict__ el,
                              const float* __restrict__ er,
                              const float* __restrict__ bias,
                              const int* __restrict__ src,
                              const int* __restrict__ rowptr,
                              OutT* __restrict__ out) {
  constexpr int C     = H * D;
  constexpr int CHUNK = 64;
  const int node = blockIdx.x;
  const int tid  = threadIdx.x;
  const int e0   = rowptr[node];
  const int e1   = rowptr[node + 1];

  __shared__ float s_alpha[CHUNK][H];
  __shared__ int   s_src[CHUNK];
  __shared__ float red[256];

  float myer[H];
#pragma unroll
  for (int h = 0; h < H; ++h) myer[h] = er[node * H + h];

  // ---- segment max per head ----
  float mx[H];
#pragma unroll
  for (int h = 0; h < H; ++h) mx[h] = -INFINITY;
  for (int e = e0 + tid; e < e1; e += blockDim.x) {
    const int s = src[e];
#pragma unroll
    for (int h = 0; h < H; ++h)
      mx[h] = fmaxf(mx[h], leaky(el[s * H + h] + myer[h]));
  }
  float m_h[H];
#pragma unroll
  for (int h = 0; h < H; ++h) {
    red[tid] = mx[h];
    __syncthreads();
    for (int st = blockDim.x >> 1; st > 0; st >>= 1) {
      if (tid < st) red[tid] = fmaxf(red[tid], red[tid + st]);
      __syncthreads();
    }
    m_h[h] = red[0];
    __syncthreads();
  }

  // ---- segment sum of exp per head ----
  float sx[H];
#pragma unroll
  for (int h = 0; h < H; ++h) sx[h] = 0.f;
  for (int e = e0 + tid; e < e1; e += blockDim.x) {
    const int s = src[e];
#pragma unroll
    for (int h = 0; h < H; ++h)
      sx[h] += expf(leaky(el[s * H + h] + myer[h]) - m_h[h]);
  }
  float den[H];
#pragma unroll
  for (int h = 0; h < H; ++h) {
    red[tid] = sx[h];
    __syncthreads();
    for (int st = blockDim.x >> 1; st > 0; st >>= 1) {
      if (tid < st) red[tid] += red[tid + st];
      __syncthreads();
    }
    den[h] = red[0];
    __syncthreads();
  }

  // ---- weighted aggregation over 64-edge chunks ----
  float acc    = 0.f;
  const int c  = tid;
  const int hc = (c < C) ? (c / D) : 0;
  for (int base = e0; base < e1; base += CHUNK) {
    const int rem = e1 - base;
    const int cnt = rem < CHUNK ? rem : CHUNK;
    if (tid < cnt) {
      const int e = base + tid;
      const int s = src[e];
      s_src[tid]  = s;
      __builtin_prefetch(fs + (size_t)s * C, 0, 0);  // global_prefetch_b8
#pragma unroll
      for (int h = 0; h < H; ++h)
        s_alpha[tid][h] =
            expf(leaky(el[s * H + h] + myer[h]) - m_h[h]) / den[h];
    }
    __syncthreads();
    if (c < C) {
      for (int j = 0; j < cnt; ++j)
        acc = fmaf(s_alpha[j][hc], fs[(size_t)s_src[j] * C + c], acc);
    }
    __syncthreads();
  }

  if (c < C) {
    float o = acc + bias[c];
    if (RELU) o = fmaxf(o, 0.f);
    out[(size_t)node * C + c] = (OutT)o;
  }
}

// ---------------------------------------------------------------------------
// Host-side orchestration (stream-ordered, graph-capture safe).
// ---------------------------------------------------------------------------
extern "C" void kernel_launch(void* const* d_in, const int* in_sizes, int n_in,
                              void* d_out, int out_size, void* d_ws,
                              size_t ws_size, hipStream_t stream) {
  const float* x   = (const float*)d_in[0];   // [50000,256]
  const float* W1  = (const float*)d_in[1];   // [256,256]
  const float* al1 = (const float*)d_in[2];   // [4,64]
  const float* ar1 = (const float*)d_in[3];   // [4,64]
  const float* b1  = (const float*)d_in[4];   // [256]
  const float* W2  = (const float*)d_in[5];   // [256,47]
  const float* al2 = (const float*)d_in[6];   // [1,47]
  const float* ar2 = (const float*)d_in[7];   // [1,47]
  const float* b2  = (const float*)d_in[8];   // [47]
  const int*   src = (const int*)d_in[9];     // [800000]
  const int*   dst = (const int*)d_in[10];    // [800000] sorted
  float*       out = (float*)d_out;           // [50000,47]

  const int N = 50000, E = 800000;
  const int K1 = 256, C1 = 256;   // layer1: H=4,D=64
  const int C2 = 47;              // layer2: H=1,D=47

  // workspace layout (~115 MB, 256B aligned chunks)
  char*  ws  = (char*)d_ws;
  size_t off = 0;
  auto alloc = [&](size_t bytes) -> void* {
    void* p = ws + off;
    off += (bytes + 255) & ~(size_t)255;
    return p;
  };
  __bf16* xb     = (__bf16*)alloc((size_t)N * K1 * sizeof(__bf16));
  __bf16* W1b    = (__bf16*)alloc((size_t)K1 * C1 * sizeof(__bf16));
  __bf16* W2b    = (__bf16*)alloc((size_t)K1 * C2 * sizeof(__bf16));
  __bf16* h1b    = (__bf16*)alloc((size_t)N * C1 * sizeof(__bf16));
  float*  fs1    = (float*)alloc((size_t)N * C1 * sizeof(float));
  float*  fs2    = (float*)alloc((size_t)N * C2 * sizeof(float));
  float*  el1    = (float*)alloc((size_t)N * 4 * sizeof(float));
  float*  er1    = (float*)alloc((size_t)N * 4 * sizeof(float));
  float*  el2    = (float*)alloc((size_t)N * sizeof(float));
  float*  er2    = (float*)alloc((size_t)N * sizeof(float));
  int*    rowptr = (int*)alloc((size_t)(N + 1) * sizeof(int));
  (void)ws_size; (void)in_sizes; (void)n_in; (void)out_size;

  // ---- bf16 conversions (one pass each; all sizes divisible by 4) ----
  {
    const int n4x = (N * K1) / 4;
    f32_to_bf16_kernel<<<(n4x + 255) / 256, 256, 0, stream>>>(x, xb, n4x);
    const int n4w1 = (K1 * C1) / 4;
    f32_to_bf16_kernel<<<(n4w1 + 255) / 256, 256, 0, stream>>>(W1, W1b, n4w1);
    const int n4w2 = (K1 * C2) / 4;
    f32_to_bf16_kernel<<<(n4w2 + 255) / 256, 256, 0, stream>>>(W2, W2b, n4w2);
  }

  const int blocksM = (N / 16 + 7) / 8;  // 391 (8 waves x 16 rows per block)

  // ---- layer 1 projection: fs1 = x @ W1  (WMMA bf16, B in LDS) ----
  {
    dim3 grid(blocksM, (C1 + 63) / 64);       // NT=4 -> 4 col groups
    size_t shmem = (size_t)4 * (K1 / 32) * 1024;  // 32 KB
    gemm_wmma_bf16<4><<<grid, 256, shmem, stream>>>(xb, W1b, fs1, N, K1, C1);
  }

  // ---- layer 1 logits ----
  attn_logits_kernel<<<(N * 4 + 255) / 256, 256, 0, stream>>>(
      fs1, al1, ar1, el1, er1, N, 4, 64);

  // ---- CSR rowptr from sorted dst (graph shared by both layers) ----
  build_rowptr_kernel<<<(E + 255) / 256, 256, 0, stream>>>(dst, rowptr, E, N);

  // ---- layer 1 softmax + aggregate + bias + relu -> h1b (bf16) ----
  gat_aggregate<4, 64, true, __bf16><<<N, 256, 0, stream>>>(
      fs1, el1, er1, b1, src, rowptr, h1b);

  // ---- layer 2 projection: fs2 = h1 @ W2 ----
  {
    dim3 grid(blocksM, 1);                    // NT=3 -> 1 col group (48>=47)
    size_t shmem = (size_t)3 * (K1 / 32) * 1024;  // 24 KB
    gemm_wmma_bf16<3><<<grid, 256, shmem, stream>>>(h1b, W2b, fs2, N, K1, C2);
  }

  // ---- layer 2 logits (H=1, D=47) ----
  attn_logits_kernel<<<(N + 255) / 256, 256, 0, stream>>>(
      fs2, al2, ar2, el2, er2, N, 1, 47);

  // ---- layer 2 softmax + aggregate + bias -> d_out [N,47]
  //      (mean over 1 head == identity) ----
  gat_aggregate<1, 47, false, float><<<N, 64, 0, stream>>>(
      fs2, el2, er2, b2, src, rowptr, out);
}
